// ChunkedAttention_64811056497157
// MI455X (gfx1250) — matrix operand order repaired and checked by reading, not compiled, not run on hardware
//
#include <hip/hip_runtime.h>

typedef __attribute__((ext_vector_type(16))) _Float16 v16h;
typedef __attribute__((ext_vector_type(8)))  float    v8f;
typedef __attribute__((ext_vector_type(4)))  float    v4f;

#define SEQ     8192
#define NH      8
#define DH      64
#define CHUNK   1024
#define NCHUNK  8
#define QTILE   128     // query rows per block (8 waves x 16 rows)
#define KSTEP   64      // keys per softmax round (2 PV fragments of 32)
#define KSTR    72      // LDS row stride in halves for K/V tiles (bank-conflict pad)
#define PSTR    68      // LDS row stride in halves for per-wave P scratch (64 + pad)

// inv_freq exponent: ln(10000)/32
#define ROPE_LOGC 0.28782313662425572f

template <bool B> struct BoolC { static constexpr bool value = B; };

__global__ __launch_bounds__(256, 1)
void chunked_attn_fa_kernel(const float* __restrict__ Q,
                            const float* __restrict__ K,
                            const float* __restrict__ V,
                            const unsigned char* __restrict__ M,
                            float* __restrict__ O)
{
    __shared__ _Float16 Klds[2][KSTEP * KSTR];
    __shared__ _Float16 Vlds[2][KSTEP * KSTR];
    __shared__ float    Madd[2][KSTEP];
    __shared__ _Float16 Plds[8][16 * PSTR];

    const float NEG_INF = -__builtin_inff();

    const int tid  = threadIdx.x;
    const int lane = tid & 31;
    const int wave = tid >> 5;
    const int hi   = (lane >= 16) ? 1 : 0;
    const int ln   = lane & 15;

    int bid = blockIdx.x;
    const int qt    = bid & 7;  bid >>= 3;   // query tile within chunk (CHUNK/QTILE = 8)
    const int h     = bid & 7;  bid >>= 3;   // head
    const int chunk = bid & 7;  bid >>= 3;   // chunk within sequence
    const int b     = bid;                   // batch

    const int qr0        = qt * QTILE + wave * 16;   // wave's first query row in chunk
    const int qpos0      = chunk * CHUNK + qr0;      // absolute position of wave row 0
    const int blockLastQ = qt * QTILE + (QTILE - 1);
    const int diagKb     = qr0 & ~(KSTEP - 1);       // the one round needing causal masks

    // staging geometry: thread owns key rows {skey, skey+32} x 8 dh (4 low + 4 high)
    const int skey = tid >> 3;            // 0..31
    const int sdq  = (tid & 7) * 4;       // low-half dh base: 0,4,...,28
    float sinvf[4];
#pragma unroll
    for (int j = 0; j < 4; ++j)
        sinvf[j] = __expf(-(float)(sdq + j) * ROPE_LOGC);

    // ---- Q fragments: two 16x32 f16 A-tiles covering Dh=64, RoPE applied ----
    v16h qfrag[2];
    {
        const int m    = ln;                  // A layout: lane%16 = M for all lanes
        const int pos  = qpos0 + m;
        const int base = ((b * SEQ + pos) * NH + h) * DH;
        const float fpos = (float)pos;
#pragma unroll
        for (int j = 0; j < 16; ++j) {
            const int p  = j >> 1;
            const int fi = ((p & 4) ? 16 : 0) + (hi ? 8 : 0) + ((p & 3) << 1) + (j & 1);
            float s, c;
            __sincosf(fpos * __expf(-(float)fi * ROPE_LOGC), &s, &c);
            const float x1 = Q[base + fi];
            const float x2 = Q[base + fi + 32];
            qfrag[0][j] = (_Float16)(x1 * c - x2 * s);   // dh = fi      (low half)
            qfrag[1][j] = (_Float16)(x1 * s + x2 * c);   // dh = fi + 32 (high half)
        }
    }

    // all-ones B fragment: P x ones gives softmax row-sums replicated in every lane
    v16h ones;
#pragma unroll
    for (int j = 0; j < 16; ++j) ones[j] = (_Float16)1.0f;

    // ---- accumulators: 4 Dv tiles + row-sum tile; running row max ----
    const v8f z8 = {0.f, 0.f, 0.f, 0.f, 0.f, 0.f, 0.f, 0.f};
    v8f o[4], osum;
#pragma unroll
    for (int d = 0; d < 4; ++d) o[d] = z8;
    osum = z8;
    float rmax[8];
#pragma unroll
    for (int r = 0; r < 8; ++r) rmax[r] = NEG_INF;

    // staging registers (software pipeline): 8x b128 per round
    v4f k1[2], k2[2], v1[2], v2[2];
    float mreg = 0.f;

    // ---- prologue: load + convert key-round 0 into buffer 0 ----
    {
#pragma unroll
        for (int g = 0; g < 2; ++g) {
            const int key   = g * 32 + skey;
            const int kbase = ((b * SEQ + chunk * CHUNK + key) * NH + h) * DH;
            k1[g] = *(const v4f*)(K + kbase + sdq);
            k2[g] = *(const v4f*)(K + kbase + sdq + 32);
            v1[g] = *(const v4f*)(V + kbase + sdq);
            v2[g] = *(const v4f*)(V + kbase + sdq + 32);
        }
        if (tid < KSTEP) mreg = M[b * SEQ + chunk * CHUNK + tid] ? 0.f : NEG_INF;
#pragma unroll
        for (int g = 0; g < 2; ++g) {
            const int key = g * 32 + skey;
            const float fpos = (float)(chunk * CHUNK + key);
#pragma unroll
            for (int j = 0; j < 4; ++j) {
                float s, c;
                __sincosf(fpos * sinvf[j], &s, &c);
                Klds[0][key * KSTR + sdq + j]      = (_Float16)(k1[g][j] * c - k2[g][j] * s);
                Klds[0][key * KSTR + sdq + 32 + j] = (_Float16)(k1[g][j] * s + k2[g][j] * c);
                Vlds[0][key * KSTR + sdq + j]      = (_Float16)v1[g][j];
                Vlds[0][key * KSTR + sdq + 32 + j] = (_Float16)v2[g][j];
            }
        }
        if (tid < KSTEP) Madd[0][tid] = mreg;
    }
    __syncthreads();

    int cur = 0;
    for (int kb = 0; kb <= blockLastQ; kb += KSTEP) {
        const int  nxt      = kb + KSTEP;
        const bool haveNext = nxt <= blockLastQ;

        // ---- issue global loads for NEXT round (latency hidden by compute) ----
        if (haveNext) {
#pragma unroll
            for (int g = 0; g < 2; ++g) {
                const int key   = g * 32 + skey;
                const int kbase = ((b * SEQ + chunk * CHUNK + nxt + key) * NH + h) * DH;
                k1[g] = *(const v4f*)(K + kbase + sdq);
                k2[g] = *(const v4f*)(K + kbase + sdq + 32);
                v1[g] = *(const v4f*)(V + kbase + sdq);
                v2[g] = *(const v4f*)(V + kbase + sdq + 32);
            }
            if (tid < KSTEP) mreg = M[b * SEQ + chunk * CHUNK + nxt + tid] ? 0.f : NEG_INF;
            // warm L2 for round kb+128 (gfx1250 global_prefetch path)
            if (nxt + KSTEP <= blockLastQ && tid < KSTEP) {
                const int nb = ((b * SEQ + chunk * CHUNK + nxt + KSTEP + tid) * NH + h) * DH;
                __builtin_prefetch(&K[nb], 0, 1);
                __builtin_prefetch(&V[nb], 0, 1);
            }
        }

        // ---- compute body: DIAG=true only for the single causal-boundary round ----
        auto compute = [&](auto diagc) {
            constexpr bool DIAG = decltype(diagc)::value;

            // four 16-key score tiles: S = Q Krope^T (contraction Dh=64, split 2x32)
            v8f acc[4];
#pragma unroll
            for (int t = 0; t < 4; ++t) acc[t] = z8;
#pragma unroll
            for (int half = 0; half < 2; ++half) {
#pragma unroll
                for (int t = 0; t < 4; ++t) {
                    v16h bfrag;     // B 32x16: (dh_local x key), lane = key col
#pragma unroll
                    for (int j = 0; j < 16; ++j) {
                        const int kk = ((j & 8) ? 16 : 0) + (hi ? 8 : 0) + (j & 7);
                        bfrag[j] = Klds[cur][(t * 16 + ln) * KSTR + half * 32 + kk];
                    }
                    acc[t] = __builtin_amdgcn_wmma_f32_16x16x32_f16(
                            false, qfrag[half], false, bfrag, (short)0, acc[t], false, false);
                }
            }

            // online softmax: one fused max-reduction over all four tiles per row
            float madd[4];
            int   kg[4];
#pragma unroll
            for (int t = 0; t < 4; ++t) {
                madd[t] = Madd[cur][t * 16 + ln];
                kg[t]   = kb + t * 16 + ln;
            }
#pragma unroll
            for (int r = 0; r < 8; ++r) {
                float s[4];
#pragma unroll
                for (int t = 0; t < 4; ++t) s[t] = acc[t][r] + madd[t];
                if (DIAG) {
                    const int qrow = qr0 + r + hi * 8;
#pragma unroll
                    for (int t = 0; t < 4; ++t)
                        s[t] = (kg[t] <= qrow) ? s[t] : NEG_INF;
                }
                float tmax = fmaxf(fmaxf(s[0], s[1]), fmaxf(s[2], s[3]));
#pragma unroll
                for (int off = 1; off < 16; off <<= 1)
                    tmax = fmaxf(tmax, __shfl_xor(tmax, off, 32));
                const float nmax  = fmaxf(rmax[r], tmax);
                const float scale = (nmax == NEG_INF) ? 1.0f : __expf(rmax[r] - nmax);
                rmax[r] = nmax;
                osum[r] *= scale;
#pragma unroll
                for (int d = 0; d < 4; ++d) o[d][r] *= scale;
#pragma unroll
                for (int t = 0; t < 4; ++t) {
                    const float p = (nmax == NEG_INF) ? 0.0f : __expf(s[t] - nmax);
                    Plds[wave][(r + hi * 8) * PSTR + t * 16 + ln] = (_Float16)p;
                }
            }

            // intra-wave cross-lane LDS RAW hazard: wait for P stores
            asm volatile("s_wait_dscnt 0x0" ::: "memory");

            // two P A-fragments (16 q-rows x 32 keys each)
            v16h pfrag[2];
#pragma unroll
            for (int f = 0; f < 2; ++f) {
#pragma unroll
                for (int j = 0; j < 16; ++j) {
                    const int p  = j >> 1;
                    const int kk = ((p & 4) ? 16 : 0) + (hi ? 8 : 0) + ((p & 3) << 1) + (j & 1);
                    pfrag[f][j] = Plds[wave][ln * PSTR + f * 32 + kk];
                }
            }

            // O += P @ V (2 key-fragments x 4 Dv tiles) and row-sums += P @ ones
#pragma unroll
            for (int f = 0; f < 2; ++f) {
#pragma unroll
                for (int d = 0; d < 4; ++d) {
                    v16h vfrag;   // B 32x16: (key x dv_col)
#pragma unroll
                    for (int j = 0; j < 16; ++j) {
                        const int kk = ((j & 8) ? 16 : 0) + (hi ? 8 : 0) + (j & 7);
                        vfrag[j] = Vlds[cur][(f * 32 + kk) * KSTR + d * 16 + ln];
                    }
                    o[d] = __builtin_amdgcn_wmma_f32_16x16x32_f16(
                            false, pfrag[f], false, vfrag, (short)0, o[d], false, false);
                }
                osum = __builtin_amdgcn_wmma_f32_16x16x32_f16(
                        false, pfrag[f], false, ones, (short)0, osum, false, false);
            }
        };

        if (kb < diagKb) {
            compute(BoolC<false>{});   // interior: no causal predication at all
        } else if (kb == diagKb) {
            compute(BoolC<true>{});    // single boundary round with causal masks
        }                              // kb > diagKb: all keys beyond rows -> skip

        // ---- RoPE/convert + store NEXT round into the other buffer ----
        if (haveNext) {
#pragma unroll
            for (int g = 0; g < 2; ++g) {
                const int key = g * 32 + skey;
                const float fpos = (float)(chunk * CHUNK + nxt + key);
#pragma unroll
                for (int j = 0; j < 4; ++j) {
                    float s, c;
                    __sincosf(fpos * sinvf[j], &s, &c);
                    Klds[cur ^ 1][key * KSTR + sdq + j]      = (_Float16)(k1[g][j] * c - k2[g][j] * s);
                    Klds[cur ^ 1][key * KSTR + sdq + 32 + j] = (_Float16)(k1[g][j] * s + k2[g][j] * c);
                    Vlds[cur ^ 1][key * KSTR + sdq + j]      = (_Float16)v1[g][j];
                    Vlds[cur ^ 1][key * KSTR + sdq + 32 + j] = (_Float16)v2[g][j];
                }
            }
            if (tid < KSTEP) Madd[cur ^ 1][tid] = mreg;
        }
        __syncthreads();
        cur ^= 1;
    }

    // ---- epilogue: normalize; row-sum is replicated in every lane of osum[r] ----
#pragma unroll
    for (int r = 0; r < 8; ++r) {
        const float inv  = (osum[r] > 0.0f) ? (1.0f / osum[r]) : 0.0f;
        const int   pos  = qpos0 + r + hi * 8;
        const int   base = ((b * SEQ + pos) * NH + h) * DH;
#pragma unroll
        for (int d = 0; d < 4; ++d) {
            O[base + d * 16 + ln] = o[d][r] * inv;
        }
    }
}

extern "C" void kernel_launch(void* const* d_in, const int* in_sizes, int n_in,
                              void* d_out, int out_size, void* d_ws, size_t ws_size,
                              hipStream_t stream) {
    const float*         Q = (const float*)d_in[0];
    const float*         K = (const float*)d_in[1];
    const float*         V = (const float*)d_in[2];
    const unsigned char* M = (const unsigned char*)d_in[3];
    float*               O = (float*)d_out;

    // blocks: b(4) x chunk(8) x head(8) x qtile(8) = 2048, 256 threads (8 wave32)
    dim3 grid(4 * NCHUNK * NH * (CHUNK / QTILE));
    chunked_attn_fa_kernel<<<grid, 256, 0, stream>>>(Q, K, V, M, O);
}
